// PolylineSubGraphLayer_82678120448523
// MI455X (gfx1250) — matrix operand order repaired and checked by reading, not compiled, not run on hardware
//
#include <hip/hip_runtime.h>

typedef __attribute__((ext_vector_type(16))) __bf16         v16bf;
typedef __attribute__((ext_vector_type(16))) unsigned short v16u;
typedef __attribute__((ext_vector_type(8)))  float          v8f;
typedef __attribute__((ext_vector_type(4)))  float          v4f;
typedef __attribute__((ext_vector_type(4)))  unsigned short v4u;

#define ROWS    64
#define NF_IN   128
#define NF_HID  256
#define NF_OUT  64
#define LNEPS   1e-5f
#define NCLUST  50000

// ---------- helpers ----------
__device__ __forceinline__ unsigned short f2bf(float f) {
  unsigned u = __float_as_uint(f);
  u += 0x7FFFu + ((u >> 16) & 1u);      // round-to-nearest-even
  return (unsigned short)(u >> 16);
}

// order-preserving float<->uint mapping for atomicMax-based segment max
__device__ __forceinline__ unsigned enc_f32(float f) {
  unsigned u = __float_as_uint(f);
  return (u & 0x80000000u) ? ~u : (u | 0x80000000u);
}
__device__ __forceinline__ float dec_f32(unsigned e) {
  unsigned u = (e & 0x80000000u) ? (e & 0x7FFFFFFFu) : ~e;
  return __uint_as_float(u);
}

// A-fragment permutation (ISA 7.12.2, 16-bit A 16x32):
// lane-half 0 holds K (rel. to tile) {0..7,16..23}, half 1 holds {8..15,24..31}
__device__ __forceinline__ int a_half(int k) { return (k >> 3) & 1; }
__device__ __forceinline__ int a_sub(int k)  { return ((k >> 4) << 3) + (k & 7); }

// one contiguous, 32B-aligned 16-half fragment load (2x b128)
__device__ __forceinline__ v16bf frag_ld(const unsigned short* p) {
  return __builtin_bit_cast(v16bf, *(const v16u*)p);
}

// ---------- kernel 0: pack weights bf16 fragment-major + segmax init ----------
// B-fragment layout: [frag][lane 0..31][j 0..15] (contiguous 32B per lane)
//   lane holds col = nt*16 + (lane&15); K = kt*32 + ((j>>3)<<4) + ((lane&16)?8:0) + (j&7)
__global__ __launch_bounds__(256) void prep_kernel(
    const float* __restrict__ W1, const float* __restrict__ W2,
    unsigned short* __restrict__ w1p, unsigned short* __restrict__ w2p,
    unsigned* __restrict__ aggr, int nW1, int nW2, int nAggr) {
  int i = blockIdx.x * 256 + threadIdx.x;
  if (i < nW1) {                        // W1: frags (kt 0..3) x (nt 0..15)
    int j = i & 15, lane = (i >> 4) & 31, fn = i >> 9;
    int kt = fn >> 4, nt = fn & 15;
    int k   = kt * 32 + ((j >> 3) << 4) + ((lane & 16) ? 8 : 0) + (j & 7);
    int col = nt * 16 + (lane & 15);
    w1p[i] = f2bf(W1[k * NF_HID + col]);
  }
  if (i < nW2) {                        // W2: frags (kt 0..7) x (nt 0..3)
    int j = i & 15, lane = (i >> 4) & 31, fn = i >> 9;
    int kt = fn >> 2, nt = fn & 3;
    int k   = kt * 32 + ((j >> 3) << 4) + ((lane & 16) ? 8 : 0) + (j & 7);
    int col = nt * 16 + (lane & 15);
    w2p[i] = f2bf(W2[k * NF_OUT + col]);
  }
  if (i < nAggr) aggr[i] = 0u;          // enc(-inf) lower bound
}

// ---------- kernel 1: fused Linear->SiLU->Linear->LayerNorm + segment max ----------
__global__ __launch_bounds__(256) void mlp_ln_kernel(
    const float* __restrict__ x, const int* __restrict__ clusters,
    const unsigned short* __restrict__ w1p, const unsigned short* __restrict__ w2p,
    const float* __restrict__ b1, const float* __restrict__ b2,
    const float* __restrict__ lnw, const float* __restrict__ lnb,
    float* __restrict__ out, unsigned* __restrict__ aggr) {
  // sH: h tile, A-fragment-permuted: [row][kt 0..7][half][16]  (32 KB)
  __shared__ __align__(32) unsigned short sH[ROWS * NF_HID];
  // sX: x tile, A-fragment-permuted: [row][kt 0..3][half][16]  (16 KB)
  // aliased after GEMM1 as f32 o tile [row][64]
  __shared__ __align__(32) unsigned short sX[ROWS * NF_IN];
  __shared__ float sMean[ROWS];
  __shared__ float sRstd[ROWS];
  float* sO = (float*)sX;

  const int tid  = threadIdx.x;
  const int lane = tid & 31;
  const int wv   = tid >> 5;             // wave id 0..7
  const int m16  = lane & 15;
  const int hbit = (lane >> 4) & 1;      // lane-half
  const int mrow = hbit ? 8 : 0;         // C/D row offset for upper half-wave
  const int row0 = blockIdx.x * ROWS;

  // stage x tile (64x128) as bf16, permuted; float4 in -> ushort4 out
#pragma unroll
  for (int i = 0; i < (ROWS * NF_IN) / (256 * 4); ++i) {
    int idx = (tid + 256 * i) * 4;                   // 4 consecutive k, same row
    int m = idx >> 7, k = idx & 127;
    int kt = k >> 5, kk = k & 31;                    // kk%8 in {0,4}: same half
    v4f xv = *(const v4f*)(x + (long)row0 * NF_IN + idx);
    v4u bv;
#pragma unroll
    for (int e = 0; e < 4; ++e) bv[e] = f2bf(xv[e]);
    *(v4u*)(sX + (((m * 4 + kt) * 2 + a_half(kk)) * 16) + a_sub(kk)) = bv;
  }
  __syncthreads();

  // ---- GEMM1: [64x128] @ [128x256], zero-init C; +b1, SiLU in epilogue ----
  {
    const int mt = wv >> 1;   // 4 row-tiles of 16
    const int nh = wv & 1;    // 2 col-halves of 128
    v8f acc[8];
#pragma unroll
    for (int nt = 0; nt < 8; ++nt)
#pragma unroll
      for (int r = 0; r < 8; ++r) acc[nt][r] = 0.0f;  // folds to inline C=0
#pragma unroll
    for (int kt = 0; kt < 4; ++kt) {
      v16bf a = frag_ld(sX + (((mt * 16 + m16) * 4 + kt) * 2 + hbit) * 16);
#pragma unroll
      for (int nt = 0; nt < 8; ++nt) {
        int fn = kt * 16 + nh * 8 + nt;
        v16bf b = frag_ld(w1p + (fn * 32 + lane) * 16);
        acc[nt] = __builtin_amdgcn_wmma_f32_16x16x32_bf16(
            false, a, false, b, (short)0, acc[nt], false, false);
      }
    }
#pragma unroll
    for (int nt = 0; nt < 8; ++nt) {
      float bias = b1[nh * 128 + nt * 16 + m16];
#pragma unroll
      for (int r = 0; r < 8; ++r) {
        int m = mt * 16 + r + mrow;
        int n = nh * 128 + nt * 16 + m16;            // GEMM2 K index
        float v = acc[nt][r] + bias;
        float h = v / (1.0f + __expf(-v));           // SiLU
        int kt2 = n >> 5, kk = n & 31;
        sH[(((m * 8 + kt2) * 2 + a_half(kk)) * 16) + a_sub(kk)] = f2bf(h);
      }
    }
  }
  __syncthreads();   // sH ready; sX dead -> reuse as sO

  // ---- GEMM2: [64x256] @ [256x64], zero-init C; +b2 in epilogue -> sO ----
  {
    const int mt = wv >> 1;
    const int np = wv & 1;    // each wave: 2 col-tiles of 16
    v8f acc[2];
#pragma unroll
    for (int c = 0; c < 2; ++c)
#pragma unroll
      for (int r = 0; r < 8; ++r) acc[c][r] = 0.0f;
#pragma unroll
    for (int kt = 0; kt < 8; ++kt) {
      v16bf a = frag_ld(sH + (((mt * 16 + m16) * 8 + kt) * 2 + hbit) * 16);
#pragma unroll
      for (int c = 0; c < 2; ++c) {
        int fn = kt * 4 + np * 2 + c;
        v16bf b = frag_ld(w2p + (fn * 32 + lane) * 16);
        acc[c] = __builtin_amdgcn_wmma_f32_16x16x32_bf16(
            false, a, false, b, (short)0, acc[c], false, false);
      }
    }
#pragma unroll
    for (int c = 0; c < 2; ++c) {
      float bias = b2[(np * 2 + c) * 16 + m16];
#pragma unroll
      for (int r = 0; r < 8; ++r) {
        int m = mt * 16 + r + mrow;
        int n = (np * 2 + c) * 16 + m16;
        sO[m * NF_OUT + n] = acc[c][r] + bias;
      }
    }
  }
  __syncthreads();

  // ---- LayerNorm stats (one thread per row) ----
  if (tid < ROWS) {
    const float* po = sO + tid * NF_OUT;
    float s = 0.f, ss = 0.f;
#pragma unroll
    for (int j = 0; j < NF_OUT; ++j) { float v = po[j]; s += v; ss += v * v; }
    float mean = s * (1.0f / NF_OUT);
    float var  = ss * (1.0f / NF_OUT) - mean * mean;
    sMean[tid] = mean;
    sRstd[tid] = rsqrtf(var + LNEPS);
  }
  __syncthreads();

  // ---- apply LN (4 cols/thread), write o to out[:, :64], feed segment max ----
#pragma unroll
  for (int i = 0; i < (ROWS * NF_OUT) / (256 * 4); ++i) {
    int idx = (tid + 256 * i) * 4;
    int r   = idx >> 6;
    int col = idx & 63;                               // multiple of 4
    v4f ov = *(const v4f*)(sO + idx);
    v4f wv4 = *(const v4f*)(lnw + col);
    v4f bv4 = *(const v4f*)(lnb + col);
    float mean = sMean[r], rstd = sRstd[r];
    long grow = (long)row0 + r;
    long cbase = (long)clusters[grow] * NF_OUT + col;
    v4f res;
#pragma unroll
    for (int e = 0; e < 4; ++e) {
      float v = (ov[e] - mean) * rstd * wv4[e] + bv4[e];
      res[e] = v;
      atomicMax(&aggr[cbase + e], enc_f32(v));
    }
    *(v4f*)(out + grow * 128 + col) = res;
  }
}

// ---------- kernel 2: gather segmax, concat, L2 normalize (1 wave / row) ----------
__global__ __launch_bounds__(256) void finalize_kernel(
    const int* __restrict__ clusters, const unsigned* __restrict__ aggr,
    float* __restrict__ out, int n) {
  int row  = (int)((blockIdx.x * 256 + threadIdx.x) >> 5);
  int lane = threadIdx.x & 31;
  if (row >= n) return;
  long base = (long)row * 128;
  int  cid  = clusters[row];
  float o0 = out[base + lane];
  float o1 = out[base + 32 + lane];
  float a0 = dec_f32(aggr[(long)cid * NF_OUT + lane]);
  float a1 = dec_f32(aggr[(long)cid * NF_OUT + 32 + lane]);
  float ss = o0 * o0 + o1 * o1 + a0 * a0 + a1 * a1;
#pragma unroll
  for (int m = 16; m >= 1; m >>= 1) ss += __shfl_xor(ss, m, 32);
  float s = 1.0f / fmaxf(sqrtf(ss), 1e-12f);
  out[base +  0 + lane] = o0 * s;
  out[base + 32 + lane] = o1 * s;
  out[base + 64 + lane] = a0 * s;
  out[base + 96 + lane] = a1 * s;
}

// ---------- launcher ----------
extern "C" void kernel_launch(void* const* d_in, const int* in_sizes, int n_in,
                              void* d_out, int out_size, void* d_ws, size_t ws_size,
                              hipStream_t stream) {
  const float* x        = (const float*)d_in[0];
  const int*   clusters = (const int*)d_in[1];
  // d_in[2] = batch (unused by the reference computation)
  const float* W1  = (const float*)d_in[3];
  const float* b1  = (const float*)d_in[4];
  const float* W2  = (const float*)d_in[5];
  const float* b2  = (const float*)d_in[6];
  const float* lnw = (const float*)d_in[7];
  const float* lnb = (const float*)d_in[8];
  float* out = (float*)d_out;

  const int N     = in_sizes[1];          // 1,000,000 rows
  const int nW1   = NF_IN * NF_HID;       // 32768
  const int nW2   = NF_HID * NF_OUT;      // 16384
  const int nAggr = NCLUST * NF_OUT;      // 3,200,000

  unsigned short* w1p  = (unsigned short*)d_ws;                              // 64 KB
  unsigned short* w2p  = w1p + nW1;                                          // 32 KB
  unsigned*       aggr = (unsigned*)((char*)d_ws + (size_t)(nW1 + nW2) * 2); // 12.8 MB

  prep_kernel<<<(nAggr + 255) / 256, 256, 0, stream>>>(W1, W2, w1p, w2p, aggr,
                                                       nW1, nW2, nAggr);
  mlp_ln_kernel<<<(N + ROWS - 1) / ROWS, 256, 0, stream>>>(
      x, clusters, w1p, w2p, b1, b2, lnw, lnb, out, aggr);
  finalize_kernel<<<(N + 7) / 8, 256, 0, stream>>>(clusters, aggr, out, N);
}
